// ConvAlloLayer_55087250539268
// MI455X (gfx1250) — matrix operand order repaired and checked by reading, not compiled, not run on hardware
//
#include <hip/hip_runtime.h>
#include <cstdint>
#include <cstddef>

// ---------------------------------------------------------------- constants
#define K_B        32
#define K_T        1024
#define K_IDIM     512
#define K_TPAD     (K_T + 4)          // 2-row zero halo on each side
#define K_NPHONES  200
#define K_MAXCNT   8
#define K_NPHONEME 100
#define K_NUMARCS  1600
#define K_KCONV    (5 * K_IDIM)       // 2560
#define K_M        (K_B * K_T)        // 32768

// GEMM tiling
#define BM 256                        // block tile M (8 waves x 32 rows)
#define BN 64                         // block tile N (4 wmma tiles)
#define BK 32                         // K-step
#define LDS_PITCH 40                  // bf16 elems per LDS row (conflict-free)

typedef __attribute__((ext_vector_type(16))) __bf16 v16bf;
typedef __attribute__((ext_vector_type(8)))  float  v8f;

struct Frag32 { uint4 lo, hi; };      // 32 bytes == one v16bf fragment
static_assert(sizeof(Frag32) == 32, "frag size");

static __device__ __forceinline__ unsigned short f2bf(float f) {
    __bf16 h = (__bf16)f;
    return __builtin_bit_cast(unsigned short, h);
}

// ------------------------------------------------- hs_pad -> bf16 with halo
__global__ __launch_bounds__(256) void pack_hs_bf16(
    const float* __restrict__ hs, unsigned short* __restrict__ dst)
{
    const long idx = (long)blockIdx.x * 256 + threadIdx.x;
    const int  i   = (int)(idx % K_IDIM);
    const long rem = idx / K_IDIM;
    const int  tt  = (int)(rem % K_TPAD);
    const int  b   = (int)(rem / K_TPAD);
    const int  t   = tt - 2;
    float v = 0.0f;
    if (t >= 0 && t < K_T) v = hs[((long)b * K_T + t) * K_IDIM + i];
    dst[idx] = f2bf(v);
}

// ------------------------- pack a row-major [N][Kd] fp32 weight matrix into
// WMMA B-fragment order: tiles (kt: K/32, nt: N/16), 32 lanes x 16 bf16 each.
// Lanes 0-15 hold K=kt*32+0..15, lanes 16-31 hold +16..31, N = nt*16+(lane&15).
__global__ __launch_bounds__(256) void pack_b_frag(
    const float* __restrict__ src, unsigned short* __restrict__ dst,
    int Kd, int nTiles)
{
    const long idx = (long)blockIdx.x * 256 + threadIdx.x;
    const int e  = (int)(idx & 15);
    const int L  = (int)((idx >> 4) & 31);
    const int nt = (int)((idx >> 9) % nTiles);
    const int kt = (int)(idx / ((long)nTiles << 9));
    const int Kc = kt * 32 + ((L >= 16) ? 16 : 0) + e;
    const int Nc = nt * 16 + (L & 15);
    dst[idx] = f2bf(src[(long)Nc * Kd + Kc]);
}

// ----------------------------------------------------------- WMMA GEMM
// C[M x N] = A(bf16) @ Bpacked + bias.  Block: 256 thr = 8 waves, tile
// BMxBN = 256x64; each wave owns 32(M)x64(N) = 2x4 WMMA tiles; K-step 32.
// Double-buffered LDS A staging via CDNA5 async global->LDS copies
// (global_load_async_to_lds_b128, ASYNCcnt): tile k+1 streams into the
// spare buffer while the 8 WMMAs of tile k execute; one barrier per K-step.
template <bool OUT_BF16>
__global__ __launch_bounds__(256) void gemm_bf16_wmma(
    const unsigned short* __restrict__ A,
    const unsigned short* __restrict__ Bp,
    const float* __restrict__ bias,
    void* __restrict__ outPtr,
    int K, int N, int convMode)
{
    __shared__ __align__(16) unsigned short ldsA[2][BM * LDS_PITCH];

    const int tid  = threadIdx.x;
    const int lane = tid & 31;
    const int wave = tid >> 5;            // 0..7 -> M sub-tile

    const int m0 = blockIdx.x * BM;
    const int n0 = blockIdx.y * BN;

    // conv mode: row n=(b,t) starts at (b*TPAD+t)*IDIM, columns contiguous
    // (covers t-2..t+2 x IDIM thanks to the halo); row stride = IDIM.
    const long baseElems = convMode
        ? ((long)(m0 >> 10) * K_TPAD + (m0 & (K_T - 1))) * (long)K_IDIM
        : (long)m0 * K;
    const int rowStride = convMode ? K_IDIM : K;

    const int nTiles = N >> 4;
    const int ntile0 = n0 >> 4;

    v8f acc[2][4];
    #pragma unroll
    for (int mi = 0; mi < 2; ++mi)
        #pragma unroll
        for (int ni = 0; ni < 4; ++ni)
            #pragma unroll
            for (int e = 0; e < 8; ++e)
                acc[mi][ni][e] = 0.0f;

    const int rlo = lane & 15;
    const int kb  = (lane < 16) ? 0 : 16;    // A-frag K base, contiguous k order (same as B packing)
    const int nk  = K >> 5;

    // Async-stage one BM x BK tile into lds[buf]; 4 x 16B per thread,
    // no destination VGPRs (pure DMA into LDS, tracked by ASYNCcnt).
    auto stageAsync = [&](int buf, int k0) {
        #pragma unroll
        for (int s = 0; s < 4; ++s) {
            const int q  = tid + (s << 8);
            const int r  = q >> 2;
            const int cc = (q & 3) << 3;
            const unsigned short* g = A + baseElems + (long)r * rowStride + k0 + cc;
            const unsigned int l =
                (unsigned int)(size_t)&ldsA[buf][r * LDS_PITCH + cc];
            asm volatile("global_load_async_to_lds_b128 %0, %1, off"
                         :: "v"(l), "v"(g) : "memory");
        }
    };

    // ---- prologue: stream tile kt=0 into lds[0]
    stageAsync(0, 0);
    asm volatile("s_wait_asynccnt 0" ::: "memory");
    __syncthreads();

    for (int kt = 0; kt < nk; ++kt) {
        const int cur = kt & 1;

        // kick off DMA of tile kt+1 into the spare buffer
        if (kt + 1 < nk) stageAsync(cur ^ 1, (kt + 1) << 5);

        // A fragments (16x32): elems 0..15 = K kb..kb+15
        v16bf af[2];
        #pragma unroll
        for (int mi = 0; mi < 2; ++mi) {
            const int rr = wave * 32 + mi * 16 + rlo;
            Frag32 f;
            f.lo = *reinterpret_cast<const uint4*>(&ldsA[cur][rr * LDS_PITCH + kb]);
            f.hi = *reinterpret_cast<const uint4*>(&ldsA[cur][rr * LDS_PITCH + kb + 8]);
            af[mi] = __builtin_bit_cast(v16bf, f);
        }

        // B fragments straight from packed global (L2-resident weights)
        v16bf bfm[4];
        #pragma unroll
        for (int ni = 0; ni < 4; ++ni) {
            const long off = ((((long)kt * nTiles) + ntile0 + ni) * 32 + lane) << 4;
            const uint4* bp4 = reinterpret_cast<const uint4*>(Bp + off);
            Frag32 f; f.lo = bp4[0]; f.hi = bp4[1];
            bfm[ni] = __builtin_bit_cast(v16bf, f);
        }

        #pragma unroll
        for (int mi = 0; mi < 2; ++mi)
            #pragma unroll
            for (int ni = 0; ni < 4; ++ni)
                acc[mi][ni] = __builtin_amdgcn_wmma_f32_16x16x32_bf16(
                    false, af[mi], false, bfm[ni],
                    (short)0, acc[mi][ni], false, false);

        // DMA of tile kt+1 must land before anyone reads it next iteration
        asm volatile("s_wait_asynccnt 0" ::: "memory");
        __syncthreads();
    }

    // store: D VGPR e -> row m_tile + e + (lane>=16 ? 8 : 0), col = lane&15
    const int roff = (lane >> 4) << 3;
    const int cl   = lane & 15;
    #pragma unroll
    for (int mi = 0; mi < 2; ++mi) {
        const size_t rbase = (size_t)m0 + wave * 32 + mi * 16 + roff;
        #pragma unroll
        for (int ni = 0; ni < 4; ++ni) {
            const int   c  = n0 + ni * 16 + cl;
            const float bv = bias[c];
            if (OUT_BF16) {
                unsigned short* colp = (unsigned short*)outPtr + rbase * N + c;
                #pragma unroll
                for (int e = 0; e < 8; ++e) { *colp = f2bf(acc[mi][ni][e] + bv); colp += N; }
            } else {
                float* colp = (float*)outPtr + rbase * N + c;
                #pragma unroll
                for (int e = 0; e < 8; ++e) { *colp = acc[mi][ni][e] + bv; colp += N; }
            }
        }
    }
}

// --------------------------------------------------------------- epilogue
// One block per (b,t) row: phone log-softmax (200), per-group (8) softmax,
// scatter exp(new_em) into 100 phoneme bins, write log(bins).
__global__ __launch_bounds__(256) void allo_epilogue(
    const float* __restrict__ Wm,              // (M, 1600)
    const float* __restrict__ phone,           // (M, 200)
    const int*   __restrict__ phonemeLabels,   // (1600)
    float* __restrict__ out)                   // (M, 100)
{
    __shared__ float spPhone[K_NPHONES];
    __shared__ float sBins[K_NPHONEME];
    __shared__ float sRed[256];

    const int  tid = threadIdx.x;
    const long row = blockIdx.x;

    float pm = -3.4e38f;
    for (int i = tid; i < K_NPHONES; i += 256) {
        const float v = phone[row * K_NPHONES + i];
        spPhone[i] = v;
        pm = fmaxf(pm, v);
    }
    sRed[tid] = pm;
    __syncthreads();
    for (int s = 128; s > 0; s >>= 1) {
        if (tid < s) sRed[tid] = fmaxf(sRed[tid], sRed[tid + s]);
        __syncthreads();
    }
    const float pmax = sRed[0];
    __syncthreads();

    float ls = 0.0f;
    for (int i = tid; i < K_NPHONES; i += 256) ls += expf(spPhone[i] - pmax);
    sRed[tid] = ls;
    __syncthreads();
    for (int s = 128; s > 0; s >>= 1) {
        if (tid < s) sRed[tid] += sRed[tid + s];
        __syncthreads();
    }
    const float psum = sRed[0];

    for (int i = tid; i < K_NPHONEME; i += 256) sBins[i] = 0.0f;
    __syncthreads();

    if (tid < K_NPHONES) {
        const int g = tid;
        const float4* wp = reinterpret_cast<const float4*>(
            Wm + row * K_NUMARCS + (long)g * K_MAXCNT);
        const float4 a = wp[0];
        const float4 b = wp[1];
        float wv[8] = {a.x, a.y, a.z, a.w, b.x, b.y, b.z, b.w};
        float gmax = wv[0];
        #pragma unroll
        for (int e = 1; e < 8; ++e) gmax = fmaxf(gmax, wv[e]);
        float gs = 0.0f;
        #pragma unroll
        for (int e = 0; e < 8; ++e) gs += expf(wv[e] - gmax);
        const float factor = expf(spPhone[g] - pmax) / (psum * gs);
        #pragma unroll
        for (int e = 0; e < 8; ++e) {
            const int ph = phonemeLabels[g * K_MAXCNT + e];
            atomicAdd(&sBins[ph], expf(wv[e] - gmax) * factor);
        }
    }
    __syncthreads();

    for (int i = tid; i < K_NPHONEME; i += 256)
        out[row * K_NPHONEME + i] = logf(sBins[i]);
}

// ------------------------------------------------------------------- host
extern "C" void kernel_launch(void* const* d_in, const int* in_sizes, int n_in,
                              void* d_out, int out_size, void* d_ws, size_t ws_size,
                              hipStream_t stream)
{
    const float* phone_out = (const float*)d_in[0];
    const float* hs_pad    = (const float*)d_in[1];
    const float* conv_w    = (const float*)d_in[2];   // (512,1,5,512) = [o][k*512+i]
    const float* conv_b    = (const float*)d_in[3];
    const float* w1        = (const float*)d_in[4];   // (512,512)
    const float* b1        = (const float*)d_in[5];
    const float* w2        = (const float*)d_in[6];   // (1600,512)
    const float* b2        = (const float*)d_in[7];
    const int*   phoneme_arc_labels = (const int*)d_in[9];
    float* out = (float*)d_out;

    char* ws = (char*)d_ws;
    size_t off = 0;
    auto carve = [&](size_t bytes) -> void* {
        void* p = ws + off;
        off = (off + bytes + 255) & ~(size_t)255;
        return p;
    };
    unsigned short* hsB = (unsigned short*)carve((size_t)K_B * K_TPAD * K_IDIM * 2);
    unsigned short* Bp1 = (unsigned short*)carve((size_t)K_KCONV * K_IDIM * 2);
    unsigned short* Bp2 = (unsigned short*)carve((size_t)K_IDIM * K_IDIM * 2);
    unsigned short* Bp3 = (unsigned short*)carve((size_t)K_IDIM * K_NUMARCS * 2);
    unsigned short* h0  = (unsigned short*)carve((size_t)K_M * K_IDIM * 2);
    unsigned short* h1  = (unsigned short*)carve((size_t)K_M * K_IDIM * 2);
    float*          Wbuf = (float*)carve((size_t)K_M * K_NUMARCS * 4);

    pack_hs_bf16<<<(K_B * K_TPAD * K_IDIM) / 256, 256, 0, stream>>>(hs_pad, hsB);
    pack_b_frag<<<(K_KCONV * K_IDIM) / 256, 256, 0, stream>>>(conv_w, Bp1, K_KCONV, K_IDIM / 16);
    pack_b_frag<<<(K_IDIM * K_IDIM) / 256, 256, 0, stream>>>(w1, Bp2, K_IDIM, K_IDIM / 16);
    pack_b_frag<<<(K_IDIM * K_NUMARCS) / 256, 256, 0, stream>>>(w2, Bp3, K_IDIM, K_NUMARCS / 16);

    // GEMM1: conv as GEMM, K=2560 (im2col via halo'd layout), out bf16 h0
    dim3 g1(K_M / BM, K_IDIM / BN);
    gemm_bf16_wmma<true><<<g1, 256, 0, stream>>>(hsB, Bp1, conv_b, h0,
                                                 K_KCONV, K_IDIM, 1);
    // GEMM2: h0 @ w1^T, out bf16 h1
    dim3 g2(K_M / BM, K_IDIM / BN);
    gemm_bf16_wmma<true><<<g2, 256, 0, stream>>>(h0, Bp2, b1, h1,
                                                 K_IDIM, K_IDIM, 0);
    // GEMM3: h1 @ w2^T, out fp32 W
    dim3 g3(K_M / BM, K_NUMARCS / BN);
    gemm_bf16_wmma<false><<<g3, 256, 0, stream>>>(h1, Bp3, b2, Wbuf,
                                                  K_IDIM, K_NUMARCS, 0);

    allo_epilogue<<<K_M, 256, 0, stream>>>(Wbuf, phone_out, phoneme_arc_labels, out);

    (void)in_sizes; (void)n_in; (void)out_size; (void)ws_size;
}
